// Attention_34857954574375
// MI455X (gfx1250) — matrix operand order repaired and checked by reading, not compiled, not run on hardware
//
#include <hip/hip_runtime.h>
#include <hip/hip_bf16.h>

#define HEADS  16
#define HDIM   64
#define EMBED  1024
#define LSEQ   2048
#define NBATCH 2
#define OUTDIM 1024
#define SCALE  0.03125f   // 1/sqrt(1024)

typedef __attribute__((ext_vector_type(16))) __bf16 v16bf;
typedef __attribute__((ext_vector_type(8)))  float  v8f;

union BF16Frag {
    v16bf v;
    __bf16 e[16];
    unsigned short u[16];
    uint4 q[2];
};

__device__ __forceinline__ v8f wmma_bf16(v16bf a, v16bf b, v8f c) {
    // D = A(16x32 bf16) * B(32x16 bf16) + C(16x16 f32)
    return __builtin_amdgcn_wmma_f32_16x16x32_bf16(false, a, false, b, (short)0, c,
                                                   false, false);
}

__device__ __forceinline__ v8f vzero8() {
    v8f z;
#pragma unroll
    for (int i = 0; i < 8; ++i) z[i] = 0.0f;
    return z;
}

// ---------------------------------------------------------------------------
// Kernel 0: one-shot f32 -> bf16 conversion of the weight matrices so the
// GEMM kernels do pure b128 loads.
// ---------------------------------------------------------------------------
__global__ __launch_bounds__(256) void convert_w_kernel(
    const float* __restrict__ Wq, const float* __restrict__ Wk,
    const float* __restrict__ Wv, const float* __restrict__ Wout,
    __bf16* __restrict__ Wq_bf, __bf16* __restrict__ Wk_bf,
    __bf16* __restrict__ Wv_bf, __bf16* __restrict__ Wout_bf)
{
    const int idx    = blockIdx.x * blockDim.x + threadIdx.x;
    const int stride = gridDim.x * blockDim.x;
    for (int i = idx; i < EMBED * OUTDIM; i += stride)
        Wout_bf[i] = (__bf16)Wout[i];
    if (idx < HDIM * HDIM) {
        Wq_bf[idx] = (__bf16)Wq[idx];
        Wk_bf[idx] = (__bf16)Wk[idx];
        Wv_bf[idx] = (__bf16)Wv[idx];
    }
}

// ---------------------------------------------------------------------------
// Kernel 1: per-head projections  y = x @ W^T  as (N*H*L) x 64 @ 64 x 64 GEMM.
// blockIdx.y selects {query/Wq/Qp, keys/Wk/Kp, values/Wv/Vp}.
// Output layout head-major: [(n*H + h)*L + l][64] bf16.
// ---------------------------------------------------------------------------
__global__ __launch_bounds__(128) void proj_kernel(
    const float* __restrict__ xq, const float* __restrict__ xk,
    const float* __restrict__ xv,
    const unsigned short* __restrict__ Wq, const unsigned short* __restrict__ Wk,
    const unsigned short* __restrict__ Wv,
    unsigned short* __restrict__ Qp, unsigned short* __restrict__ Kp,
    unsigned short* __restrict__ Vp)
{
    const int lane = threadIdx.x & 31;
    const int wv   = threadIdx.x >> 5;
    const int n16  = lane & 15;
    const int hi   = lane >> 4;

    const float* src; const unsigned short* W; unsigned short* dst;
    if (blockIdx.y == 0)      { src = xq; W = Wq; dst = Qp; }
    else if (blockIdx.y == 1) { src = xk; W = Wk; dst = Kp; }
    else                      { src = xv; W = Wv; dst = Vp; }

    const int rid0 = (blockIdx.x * 4 + wv) * 16;    // head-major flat row base
    const int rid  = rid0 + n16;                    // this lane's A-matrix row
    const int l  = rid & (LSEQ - 1);
    const int h  = (rid / LSEQ) & (HEADS - 1);
    const int nb = rid / (LSEQ * HEADS);
    const float* xrow = src + (((size_t)nb * LSEQ + l) * HEADS + h) * HDIM;

    // A fragments: 16x32 bf16, two K-steps over d=0..63 (native v_cvt_pk)
    BF16Frag A[2];
#pragma unroll
    for (int s = 0; s < 2; ++s) {
        const float* p0 = xrow + s * 32 + hi * 8;
        const float* p1 = xrow + s * 32 + 16 + hi * 8;
#pragma unroll
        for (int i = 0; i < 8; ++i) {
            A[s].e[i]     = (__bf16)p0[i];
            A[s].e[8 + i] = (__bf16)p1[i];
        }
    }

    v8f acc[4];
#pragma unroll
    for (int t = 0; t < 4; ++t) acc[t] = vzero8();

#pragma unroll
    for (int s = 0; s < 2; ++s) {
#pragma unroll
        for (int t = 0; t < 4; ++t) {
            // B[k=d][n=e] = W[e0+n][d]; two 8-bf16 runs per lane (table order)
            const unsigned short* wp =
                W + (size_t)(t * 16 + n16) * HDIM + s * 32 + hi * 8;
            BF16Frag B;
            B.q[0] = *(const uint4*)wp;
            B.q[1] = *(const uint4*)(wp + 16);
            acc[t] = wmma_bf16(A[s].v, B.v, acc[t]);
        }
    }

    __bf16* dstb = (__bf16*)dst;
#pragma unroll
    for (int t = 0; t < 4; ++t)
#pragma unroll
        for (int r = 0; r < 8; ++r)
            dstb[(size_t)(rid0 + hi * 8 + r) * HDIM + t * 16 + n16] =
                (__bf16)acc[t][r];
}

// ---------------------------------------------------------------------------
// Kernel 2: flash attention, 64-key chunks. Block = 4 waves sharing (n,h);
// wave owns 16 q rows. Per chunk: 8 WMMAs for S (K fragments straight from
// global), online softmax (wave16 shuffles, amortized over 64 keys), P via
// LDS for the D->A relayout, V staged transposed in LDS so P@V B-fragments
// are contiguous b128 loads; 8 WMMAs for O += P@V.
// ---------------------------------------------------------------------------
__global__ __launch_bounds__(128) void flash_kernel(
    const unsigned short* __restrict__ Qp, const unsigned short* __restrict__ Kp,
    const unsigned short* __restrict__ Vp, unsigned short* __restrict__ attn)
{
    __shared__ __align__(16) unsigned short vbufT[HDIM * 64];   // [d][key] bf16
    __shared__ __align__(16) unsigned short pbuf[4][16 * 64];   // per-wave P tile

    const int lane = threadIdx.x & 31;
    const int wv   = threadIdx.x >> 5;
    const int n16  = lane & 15;
    const int hi   = lane >> 4;

    const int qblk = blockIdx.x & 31;            // L/64 = 32
    const int h    = (blockIdx.x >> 5) & 15;
    const int nb   = blockIdx.x >> 9;
    const size_t hbase = (size_t)(nb * HEADS + h) * LSEQ * HDIM;
    const int q0 = qblk * 64 + wv * 16;

    // Q tile 16x64 as two A fragments (resident whole kernel)
    BF16Frag aQ[2];
    {
        const unsigned short* qrow = Qp + hbase + (size_t)(q0 + n16) * HDIM;
#pragma unroll
        for (int s = 0; s < 2; ++s) {
            aQ[s].q[0] = *(const uint4*)(qrow + s * 32 + hi * 8);
            aQ[s].q[1] = *(const uint4*)(qrow + s * 32 + 16 + hi * 8);
        }
    }

    float mrow[8], lrow[8];
    v8f acc[4];
#pragma unroll
    for (int r = 0; r < 8; ++r) { mrow[r] = -1.0e30f; lrow[r] = 0.0f; }
#pragma unroll
    for (int t = 0; t < 4; ++t) acc[t] = vzero8();

    const int skey = threadIdx.x & 31;   // staging role: key row (+0 / +32)
    const int dseg = threadIdx.x >> 5;   // staging role: 16-dim segment
    __bf16* pb = (__bf16*)pbuf[wv];

    for (int j0 = 0; j0 < LSEQ; j0 += 64) {
        __syncthreads();                 // previous chunk's V reads done
        // ---- stage V chunk transposed: vbufT[d][key], 64 keys ----
#pragma unroll
        for (int kk = 0; kk < 64; kk += 32) {
            const unsigned short* vrow =
                Vp + hbase + (size_t)(j0 + kk + skey) * HDIM + dseg * 16;
            BF16Frag tv;
            tv.q[0] = *(const uint4*)(vrow);
            tv.q[1] = *(const uint4*)(vrow + 8);
#pragma unroll
            for (int i = 0; i < 16; ++i)
                vbufT[(dseg * 16 + i) * 64 + kk + skey] = tv.u[i];
        }
        if (j0 + 64 < LSEQ)
            __builtin_prefetch(Kp + hbase + (size_t)(j0 + 64 + lane * 2) * HDIM, 0, 1);
        __syncthreads();                 // V staged

        // ---- scores: four 16x16 tiles over this 64-key chunk ----
        v8f sc[4];
#pragma unroll
        for (int t = 0; t < 4; ++t) sc[t] = vzero8();
#pragma unroll
        for (int s = 0; s < 2; ++s) {
#pragma unroll
            for (int t = 0; t < 4; ++t) {
                // B[k=d][n=keycol] = K[j0+16t+n][d]; two 8-half runs (table order)
                const unsigned short* kr = Kp + hbase +
                    (size_t)(j0 + t * 16 + n16) * HDIM + s * 32 + hi * 8;
                BF16Frag b;
                b.q[0] = *(const uint4*)kr;
                b.q[1] = *(const uint4*)(kr + 16);
                sc[t] = wmma_bf16(aQ[s].v, b.v, sc[t]);
            }
        }

        // ---- online softmax (rows = 8*hi + r, cols across 16 lanes) ----
#pragma unroll
        for (int r = 0; r < 8; ++r) {
            float a0 = sc[0][r] * SCALE;
            float a1 = sc[1][r] * SCALE;
            float a2 = sc[2][r] * SCALE;
            float a3 = sc[3][r] * SCALE;
            float mc = fmaxf(fmaxf(a0, a1), fmaxf(a2, a3));
#pragma unroll
            for (int off = 8; off >= 1; off >>= 1)
                mc = fmaxf(mc, __shfl_xor(mc, off, 16));
            float mn = fmaxf(mrow[r], mc);
            float rs = __expf(mrow[r] - mn);
            float p0 = __expf(a0 - mn);
            float p1 = __expf(a1 - mn);
            float p2 = __expf(a2 - mn);
            float p3 = __expf(a3 - mn);
            float ls = (p0 + p1) + (p2 + p3);
#pragma unroll
            for (int off = 8; off >= 1; off >>= 1)
                ls += __shfl_xor(ls, off, 16);
            mrow[r] = mn;
            lrow[r] = lrow[r] * rs + ls;
            acc[0][r] *= rs; acc[1][r] *= rs; acc[2][r] *= rs; acc[3][r] *= rs;
            const int prow = (hi * 8 + r) * 64;
            pb[prow + n16]      = (__bf16)p0;
            pb[prow + 16 + n16] = (__bf16)p1;
            pb[prow + 32 + n16] = (__bf16)p2;
            pb[prow + 48 + n16] = (__bf16)p3;
        }
        asm volatile("s_wait_dscnt 0" ::: "memory");  // own P stores -> loads

        // ---- O += P(16x64) @ V(64x64): 2 K-steps x 4 dim-tiles ----
#pragma unroll
        for (int s = 0; s < 2; ++s) {
            BF16Frag aP;
            aP.q[0] = *(const uint4*)&pbuf[wv][n16 * 64 + s * 32 + hi * 8];
            aP.q[1] = *(const uint4*)&pbuf[wv][n16 * 64 + s * 32 + 16 + hi * 8];
#pragma unroll
            for (int t = 0; t < 4; ++t) {
                const unsigned short* vp =
                    &vbufT[(t * 16 + n16) * 64 + s * 32 + hi * 8];
                BF16Frag bV;
                bV.q[0] = *(const uint4*)vp;
                bV.q[1] = *(const uint4*)(vp + 16);
                acc[t] = wmma_bf16(aP.v, bV.v, acc[t]);
            }
        }
    }

    // ---- epilogue: normalize, write bf16 attn output [n][q][h*64+d] ----
    __bf16* attnb = (__bf16*)attn;
#pragma unroll
    for (int r = 0; r < 8; ++r) {
        const float inv = 1.0f / lrow[r];
        const int qg = q0 + hi * 8 + r;
        const size_t obase = ((size_t)nb * LSEQ + qg) * EMBED + h * HDIM;
#pragma unroll
        for (int t = 0; t < 4; ++t)
            attnb[obase + t * 16 + n16] = (__bf16)(acc[t][r] * inv);
    }
}

// ---------------------------------------------------------------------------
// Kernel 3: out = attn(4096x1024 bf16) @ Wout_bf^T + bout.
// One wave per 16x64 output strip: A fragment loaded once per k-step and
// reused by 4 WMMAs (4 N-tiles); 128 WMMAs per wave over K=1024.
// ---------------------------------------------------------------------------
__global__ __launch_bounds__(128) void outproj_kernel(
    const unsigned short* __restrict__ attn, const unsigned short* __restrict__ Wout,
    const float* __restrict__ bout, float* __restrict__ out)
{
    const int lane = threadIdx.x & 31;
    const int wv   = threadIdx.x >> 5;
    const int n16  = lane & 15;
    const int hi   = lane >> 4;

    const int g    = blockIdx.x * 4 + wv;
    const int mt   = g >> 4;         // 16 strip groups across 1024 outputs
    const int ng   = g & 15;
    const int row0 = mt * 16;
    const int o0   = ng * 64;

    v8f acc[4];
#pragma unroll
    for (int t = 0; t < 4; ++t) acc[t] = vzero8();

    const unsigned short* ar = attn + (size_t)(row0 + n16) * EMBED;

    for (int k = 0; k < EMBED; k += 32) {
        BF16Frag A;
        A.q[0] = *(const uint4*)(ar + k + hi * 8);
        A.q[1] = *(const uint4*)(ar + k + 16 + hi * 8);
#pragma unroll
        for (int t = 0; t < 4; ++t) {
            const unsigned short* bp =
                Wout + (size_t)(o0 + t * 16 + n16) * EMBED + k + hi * 8;
            BF16Frag B;
            B.q[0] = *(const uint4*)bp;
            B.q[1] = *(const uint4*)(bp + 16);
            acc[t] = wmma_bf16(A.v, B.v, acc[t]);
        }
    }

#pragma unroll
    for (int t = 0; t < 4; ++t) {
        const float bb = bout[o0 + t * 16 + n16];
#pragma unroll
        for (int r = 0; r < 8; ++r)
            out[(size_t)(row0 + hi * 8 + r) * OUTDIM + o0 + t * 16 + n16] =
                acc[t][r] + bb;
    }
}

// ---------------------------------------------------------------------------
extern "C" void kernel_launch(void* const* d_in, const int* in_sizes, int n_in,
                              void* d_out, int out_size, void* d_ws, size_t ws_size,
                              hipStream_t stream) {
    const float* values = (const float*)d_in[0];
    const float* keys   = (const float*)d_in[1];
    const float* query  = (const float*)d_in[2];
    const float* Wv     = (const float*)d_in[3];
    const float* Wk     = (const float*)d_in[4];
    const float* Wq     = (const float*)d_in[5];
    const float* Wout   = (const float*)d_in[6];
    const float* bout   = (const float*)d_in[7];
    float* out = (float*)d_out;

    const size_t PROJ = (size_t)NBATCH * HEADS * LSEQ * HDIM;   // 4,194,304 elems
    unsigned short* Qp      = (unsigned short*)d_ws;
    unsigned short* Kp      = Qp + PROJ;
    unsigned short* Vp      = Kp + PROJ;
    unsigned short* attn    = Vp + PROJ;
    unsigned short* Wout_bf = attn + PROJ;                      // 1M halves
    unsigned short* Wq_bf   = Wout_bf + (size_t)EMBED * OUTDIM;
    unsigned short* Wk_bf   = Wq_bf + HDIM * HDIM;
    unsigned short* Wv_bf   = Wk_bf + HDIM * HDIM;              // total ~34 MB

    // 0) weights -> bf16 once
    convert_w_kernel<<<1024, 256, 0, stream>>>(Wq, Wk, Wv, Wout,
                                               (__bf16*)Wq_bf, (__bf16*)Wk_bf,
                                               (__bf16*)Wv_bf, (__bf16*)Wout_bf);
    // 1) projections: (N*H*L)/16 = 4096 row-tiles, 4 waves/block
    proj_kernel<<<dim3(1024, 3), 128, 0, stream>>>(query, keys, values,
                                                   Wq_bf, Wk_bf, Wv_bf,
                                                   Qp, Kp, Vp);
    // 2) flash attention: N*H*(L/64) = 1024 blocks of 4 waves
    flash_kernel<<<1024, 128, 0, stream>>>(Qp, Kp, Vp, attn);
    // 3) output projection: 256 row-tiles x 16 strips / 4 per block
    outproj_kernel<<<1024, 128, 0, stream>>>(attn, Wout_bf, bout, out);
}
